// VQVAE_56453050138865
// MI455X (gfx1250) — compile-verified, matrix operand order checked by reading, not run on hardware
//
#include <hip/hip_runtime.h>

// ---------------------------------------------------------------------------
// VQ-VAE forward for MI455X (gfx1250): bf16 WMMA GEMMs (double-buffered LDS,
// 64x64 wave tiles) + fused logsumexp similarity kernel
// ---------------------------------------------------------------------------

typedef unsigned short u16;
typedef __attribute__((ext_vector_type(16))) __bf16 v16bf;
typedef __attribute__((ext_vector_type(8)))  float  v8f;

union Frag16 { v16bf v; uint4 q[2]; };

// problem sizes
constexpr int N_  = 32768;
constexpr int IN_ = 768;
constexpr int H1_ = 2048;
constexpr int H2_ = 1024;
constexpr int E_  = 64;
constexpr int NE_ = 256;
constexpr int KP_ = 4096;
constexpr int TT_ = 8192;
constexpr float TEMP_   = 0.1f;
constexpr float MARGIN_ = 0.2f;
constexpr float BETA_   = 0.001f;

// d_out layout (float elements, concatenated return order)
constexpr size_t OUT_OFF   = 0;
constexpr size_t VQ_OFF    = (size_t)N_ * IN_;
constexpr size_t IDX_OFF   = VQ_OFF + 1;
constexpr size_t XQ_OFF    = IDX_OFF + N_;
constexpr size_t OUTER_OFF = XQ_OFF + (size_t)N_ * E_;
constexpr size_t TRIP_OFF  = OUTER_OFF + 1;
constexpr size_t QD_OFF    = TRIP_OFF + 1;

// d_ws layout (bytes)
constexpr size_t XB_OFF  = 0;                                  // N*IN bf16 (x, then q_embs)
constexpr size_t H1B_OFF = XB_OFF  + (size_t)N_ * IN_ * 2;     // N*H1 bf16
constexpr size_t H2B_OFF = H1B_OFF + (size_t)N_ * H1_ * 2;     // N*H2 bf16
constexpr size_t ZF_OFF  = H2B_OFF + (size_t)N_ * H2_ * 2;     // N*E f32
constexpr size_t QZF_OFF = ZF_OFF  + (size_t)N_ * E_ * 4;      // N*E f32
constexpr size_t XQB_OFF = QZF_OFF + (size_t)N_ * E_ * 4;      // N*E bf16
constexpr size_t XNB_OFF = XQB_OFF + (size_t)N_ * E_ * 2;      // N*E bf16
constexpr size_t W0T_OFF = XNB_OFF + (size_t)N_ * E_ * 2;      // H1*IN bf16
constexpr size_t W1T_OFF = W0T_OFF + (size_t)H1_ * IN_ * 2;    // H2*H1 bf16
constexpr size_t W2T_OFF = W1T_OFF + (size_t)H2_ * H1_ * 2;    // E*H2 bf16
constexpr size_t D0T_OFF = W2T_OFF + (size_t)E_ * H2_ * 2;     // H2*E bf16
constexpr size_t D1T_OFF = D0T_OFF + (size_t)H2_ * E_ * 2;     // H1*H2 bf16
constexpr size_t D2T_OFF = D1T_OFF + (size_t)H1_ * H2_ * 2;    // IN*H1 bf16
constexpr size_t LSE_OFF = D2T_OFF + (size_t)IN_ * H1_ * 2;    // KP f32
constexpr size_t POS_OFF = LSE_OFF + (size_t)KP_ * 4;          // KP f32

__device__ __forceinline__ u16 f2bf(float f) {
  unsigned int u = __float_as_uint(f);
  unsigned int r = u + 0x7FFFu + ((u >> 16) & 1u);   // round-to-nearest-even
  return (u16)(r >> 16);
}
__device__ __forceinline__ float bf2f(u16 h) {
  return __uint_as_float(((unsigned int)h) << 16);
}
__device__ __forceinline__ uint4 lds_ld8h(const u16* p) {
  uint2 a = *reinterpret_cast<const uint2*>(p);
  uint2 b = *reinterpret_cast<const uint2*>(p + 4);
  return make_uint4(a.x, a.y, b.x, b.y);
}
__device__ __forceinline__ void lds_st16h(u16* p, uint4 a, uint4 b) {
  reinterpret_cast<uint2*>(p)[0]      = make_uint2(a.x, a.y);
  reinterpret_cast<uint2*>(p + 4)[0]  = make_uint2(a.z, a.w);
  reinterpret_cast<uint2*>(p + 8)[0]  = make_uint2(b.x, b.y);
  reinterpret_cast<uint2*>(p + 12)[0] = make_uint2(b.z, b.w);
}
__device__ __forceinline__ uint4 gld(const u16* p) {
  return *reinterpret_cast<const uint4*>(p);
}

// ------------------------- utility kernels ---------------------------------

__global__ void k_init_scalars(float* dout) {
  if (threadIdx.x == 0) {
    dout[VQ_OFF]    = 0.f;
    dout[OUTER_OFF] = 0.f;
    dout[TRIP_OFF]  = 0.f;
    dout[QD_OFF]    = 1.f;   // qd_align_loss = 1 - mean(w*cos)
  }
}

__global__ void k_f32_to_bf16(const float* __restrict__ s, u16* __restrict__ d, int n) {
  for (int i = blockIdx.x * blockDim.x + threadIdx.x; i < n; i += gridDim.x * blockDim.x)
    d[i] = f2bf(s[i]);
}

// D[c*R + r] = bf16(S[r*C + c])   (W[K][N] -> Wt[N][K])
__global__ void k_transpose_bf16(const float* __restrict__ S, u16* __restrict__ D,
                                 int R, int C) {
  int c = blockIdx.x * 16 + threadIdx.x;
  int r = blockIdx.y * 16 + threadIdx.y;
  if (r < R && c < C) D[(size_t)c * R + r] = f2bf(S[(size_t)r * C + c]);
}

// ------------------------- WMMA GEMM ---------------------------------------
// C[M,Nn] = A[M,K](bf16, row-major) @ Bt[Nn,K]^T (+bias, optional relu)
// Workgroup tile 128x256, 8 waves (2M x 4N), wave tile 64x64 = 4x4 WMMA
// subtiles. Double-buffered LDS: global loads for tile k+1 issue before the
// WMMA block of tile k, stores land after it -> latency hidden by compute.
__global__ __launch_bounds__(256) void k_gemm_bf16(
    const u16* __restrict__ A, const u16* __restrict__ Bt,
    const float* __restrict__ bias,
    float* __restrict__ outF, u16* __restrict__ outB,
    int M, int Nn, int K, int relu)
{
  __shared__ __align__(16) u16 As[2][128 * 40];
  __shared__ __align__(16) u16 Bs[2][256 * 40];

  const int tid  = threadIdx.x;
  const int lane = tid & 31;
  const int wave = tid >> 5;
  const int wm = (wave >> 2) * 64;     // 0 / 64
  const int wn = (wave & 3) * 64;      // 0,64,128,192
  const int m0 = blockIdx.y * 128;
  const int n0 = blockIdx.x * 256;

  const int hl   = lane >> 4;          // wave half
  const int mr   = lane & 15;          // row within 16x16 subtile
  const int koff = hl * 8;             // ISA 16-bit A/B fragment K offset

  const int srow = tid >> 1;           // staging row 0..127
  const int scol = (tid & 1) * 16;     // staging halfword offset 0/16

  const u16* gA = A + (size_t)(m0 + srow) * K + scol;
  const int brow0 = n0 + srow;
  const int brow1 = n0 + srow + 128;
  const u16* gB0 = Bt + (size_t)brow0 * K + scol;
  const u16* gB1 = Bt + (size_t)brow1 * K + scol;
  const bool bok0 = brow0 < Nn;
  const bool bok1 = brow1 < Nn;
  const uint4 z4 = make_uint4(0, 0, 0, 0);

  v8f acc[4][4];
#pragma unroll
  for (int s = 0; s < 4; s++)
#pragma unroll
    for (int t = 0; t < 4; t++)
#pragma unroll
      for (int r = 0; r < 8; r++) acc[s][t][r] = 0.f;

  // ---- prologue: stage K-tile 0 into buffer 0 ----
  uint4 ra0 = gld(gA),            ra1 = gld(gA + 8);
  uint4 rb00 = bok0 ? gld(gB0) : z4, rb01 = bok0 ? gld(gB0 + 8) : z4;
  uint4 rb10 = bok1 ? gld(gB1) : z4, rb11 = bok1 ? gld(gB1 + 8) : z4;
  lds_st16h(&As[0][srow * 40 + scol], ra0, ra1);
  lds_st16h(&Bs[0][srow * 40 + scol], rb00, rb01);
  lds_st16h(&Bs[0][(srow + 128) * 40 + scol], rb10, rb11);
  __syncthreads();

  for (int k0 = 0; k0 < K; k0 += 32) {
    const int cur = (k0 >> 5) & 1;
    const bool nxt = (k0 + 32) < K;

    // issue global loads for the next K-tile (latency overlapped with WMMA)
    if (nxt) {
      const int kn = k0 + 32;
      ra0 = gld(gA + kn); ra1 = gld(gA + kn + 8);
      rb00 = bok0 ? gld(gB0 + kn) : z4; rb01 = bok0 ? gld(gB0 + kn + 8) : z4;
      rb10 = bok1 ? gld(gB1 + kn) : z4; rb11 = bok1 ? gld(gB1 + kn + 8) : z4;
      if (kn + 32 < K) __builtin_prefetch(gA + kn + 32, 0, 3);
    }

    // ---- compute current tile: 8 fragment loads, 16 WMMAs ----
    Frag16 fa[4], fb[4];
#pragma unroll
    for (int s = 0; s < 4; s++) {
      const u16* p = &As[cur][(wm + s * 16 + mr) * 40];
      fa[s].q[0] = lds_ld8h(p + koff);
      fa[s].q[1] = lds_ld8h(p + 16 + koff);
    }
#pragma unroll
    for (int t = 0; t < 4; t++) {
      const u16* p = &Bs[cur][(wn + t * 16 + mr) * 40];
      fb[t].q[0] = lds_ld8h(p + koff);
      fb[t].q[1] = lds_ld8h(p + 16 + koff);
    }
#pragma unroll
    for (int s = 0; s < 4; s++)
#pragma unroll
      for (int t = 0; t < 4; t++)
        acc[s][t] = __builtin_amdgcn_wmma_f32_16x16x32_bf16(
            false, fa[s].v, false, fb[t].v, (short)0, acc[s][t], false, false);

    // ---- write next tile into the other buffer ----
    if (nxt) {
      const int nb = cur ^ 1;
      lds_st16h(&As[nb][srow * 40 + scol], ra0, ra1);
      lds_st16h(&Bs[nb][srow * 40 + scol], rb00, rb01);
      lds_st16h(&Bs[nb][(srow + 128) * 40 + scol], rb10, rb11);
    }
    __syncthreads();
  }

  // ---- epilogue: bias + relu, store f32 and/or bf16 ----
#pragma unroll
  for (int t = 0; t < 4; t++) {
    const int col = n0 + wn + t * 16 + mr;
    if (col >= Nn) continue;
    const float bv = bias ? bias[col] : 0.f;
#pragma unroll
    for (int s = 0; s < 4; s++) {
      const int rbase = m0 + wm + s * 16 + (hl ? 8 : 0);
#pragma unroll
      for (int r = 0; r < 8; r++) {
        float v = acc[s][t][r] + bv;
        if (relu) v = fmaxf(v, 0.f);
        const size_t o = (size_t)(rbase + r) * Nn + col;
        if (outF) outF[o] = v;
        if (outB) outB[o] = f2bf(v);
      }
    }
  }
}

// ------------------------- VQ argmin + losses ------------------------------
__global__ __launch_bounds__(256) void k_vq(const float* __restrict__ z,
                                            const float* __restrict__ cb,
                                            float* __restrict__ dout,
                                            u16* __restrict__ xqb,
                                            u16* __restrict__ xnb)
{
  __shared__ float cbs[NE_ * E_];
  __shared__ float c2[NE_];
  const int tid = threadIdx.x;
  {
    float s2 = 0.f;
    for (int j = 0; j < E_; j++) {
      float v = cb[(size_t)tid * E_ + j];
      cbs[tid * E_ + j] = v;
      s2 += v * v;
    }
    c2[tid] = s2;
  }
  __syncthreads();

  const int i = blockIdx.x * 256 + tid;
  float zr[E_];
#pragma unroll
  for (int j = 0; j < E_; j++) zr[j] = z[(size_t)i * E_ + j];

  int best = 0;
  float bd = 3.4e38f;
  for (int c = 0; c < NE_; c++) {
    float d = 0.f;
#pragma unroll
    for (int j = 0; j < E_; j++) d += cbs[c * E_ + j] * zr[j];
    d = c2[c] - 2.f * d;
    if (d < bd) { bd = d; best = c; }
  }

  float diff2 = 0.f, n2 = 0.f;
#pragma unroll
  for (int j = 0; j < E_; j++) {
    const float xq = cbs[best * E_ + j];
    dout[XQ_OFF + (size_t)i * E_ + j] = xq;        // x_q_st forward value == x_q
    xqb[(size_t)i * E_ + j] = f2bf(xq);
    const float df = xq - zr[j];
    diff2 += df * df;
    n2 += xq * xq;
  }
  const float inv = 1.f / fmaxf(sqrtf(n2), 1e-12f);
#pragma unroll
  for (int j = 0; j < E_; j++)
    xnb[(size_t)i * E_ + j] = f2bf(cbs[best * E_ + j] * inv);

  dout[IDX_OFF + i] = (float)best;
  atomicAdd(&dout[VQ_OFF], diff2 * ((1.f + BETA_) / ((float)N_ * (float)E_)));
}

// ------------------------- small loss kernels ------------------------------
__global__ void k_cos(const float* __restrict__ z, const float* __restrict__ qz,
                      const float* __restrict__ w, float* __restrict__ dout) {
  const int i = blockIdx.x * blockDim.x + threadIdx.x;
  if (i >= N_) return;
  float d = 0.f, a = 0.f, b = 0.f;
#pragma unroll
  for (int j = 0; j < E_; j++) {
    const float zv = z[(size_t)i * E_ + j];
    const float qv = qz[(size_t)i * E_ + j];
    d += zv * qv; a += zv * zv; b += qv * qv;
  }
  const float cs = d / fmaxf(sqrtf(a) * sqrtf(b), 1e-8f);
  atomicAdd(&dout[QD_OFF], -w[i] * cs / (float)N_);
}

__global__ void k_trip(const float* __restrict__ z, const int* __restrict__ tr,
                       float* __restrict__ dout) {
  const int i = blockIdx.x * blockDim.x + threadIdx.x;
  if (i >= TT_) return;
  const int ia = tr[3 * i], ip = tr[3 * i + 1], in = tr[3 * i + 2];
  float dp = 0.f, dn = 0.f;
#pragma unroll
  for (int j = 0; j < E_; j++) {
    const float av = z[(size_t)ia * E_ + j];
    const float pv = z[(size_t)ip * E_ + j];
    const float nv = z[(size_t)in * E_ + j];
    dp += (av - pv) * (av - pv);
    dn += (av - nv) * (av - nv);
  }
  dp = sqrtf(dp + 1e-12f);
  dn = sqrtf(dn + 1e-12f);
  const float l = fmaxf(dp - dn + MARGIN_, 0.f);
  atomicAdd(&dout[TRIP_OFF], l / (float)TT_);
}

__global__ void k_pos(const u16* __restrict__ xn, const int* __restrict__ pairs,
                      float* __restrict__ pos) {
  const int i = blockIdx.x * blockDim.x + threadIdx.x;
  if (i >= KP_) return;
  const int p0 = pairs[2 * i], p1 = pairs[2 * i + 1];
  float d = 0.f;
#pragma unroll
  for (int j = 0; j < E_; j++)
    d += bf2f(xn[(size_t)p0 * E_ + j]) * bf2f(xn[(size_t)p1 * E_ + j]);
  pos[i] = d * (1.f / TEMP_);
}

// ------------------------- sim + fused logsumexp ---------------------------
// Each wave: 16 anchor rows (gathered via pairs[:,0]) vs all N columns.
// WMMA 16x16x32 (2 per 16-col tile, K=64); online (max,sum) LSE per slot.
__global__ __launch_bounds__(256) void k_sim(const u16* __restrict__ xn,
                                             const int* __restrict__ pairs,
                                             float* __restrict__ lse, int Ntot)
{
  const int lane = threadIdx.x & 31;
  const int wave = threadIdx.x >> 5;
  const int a0 = (blockIdx.x * 8 + wave) * 16;
  const int hl = lane >> 4;
  const int mr = lane & 15;
  const int koff = hl * 8;

  const int arow = pairs[2 * (a0 + mr)];
  Frag16 fa[2];
#pragma unroll
  for (int s = 0; s < 2; s++) {
    const u16* p = xn + (size_t)arow * E_ + s * 32;
    fa[s].q[0] = *reinterpret_cast<const uint4*>(p + koff);
    fa[s].q[1] = *reinterpret_cast<const uint4*>(p + 16 + koff);
  }

  float mx[8], sm[8];
#pragma unroll
  for (int r = 0; r < 8; r++) { mx[r] = -3.0e38f; sm[r] = 0.f; }

  for (int nt = 0; nt < Ntot; nt += 16) {
    Frag16 fb[2];
    const u16* p = xn + (size_t)(nt + mr) * E_;
#pragma unroll
    for (int s = 0; s < 2; s++) {
      fb[s].q[0] = *reinterpret_cast<const uint4*>(p + s * 32 + koff);
      fb[s].q[1] = *reinterpret_cast<const uint4*>(p + s * 32 + 16 + koff);
    }
    v8f acc;
#pragma unroll
    for (int r = 0; r < 8; r++) acc[r] = 0.f;
    acc = __builtin_amdgcn_wmma_f32_16x16x32_bf16(false, fa[0].v, false, fb[0].v,
                                                  (short)0, acc, false, false);
    acc = __builtin_amdgcn_wmma_f32_16x16x32_bf16(false, fa[1].v, false, fb[1].v,
                                                  (short)0, acc, false, false);
#pragma unroll
    for (int r = 0; r < 8; r++) {
      const float v = acc[r] * (1.f / TEMP_);
      const float mo = mx[r];
      const float mn = fmaxf(mo, v);
      sm[r] = sm[r] * __expf(mo - mn) + __expf(v - mn);
      mx[r] = mn;
    }
  }

  // merge across the 16 column-lanes of each half (xor 1,2,4,8 stays in-half)
#pragma unroll
  for (int off = 1; off < 16; off <<= 1) {
#pragma unroll
    for (int r = 0; r < 8; r++) {
      const float om = __shfl_xor(mx[r], off, 32);
      const float os = __shfl_xor(sm[r], off, 32);
      const float mn = fmaxf(mx[r], om);
      sm[r] = sm[r] * __expf(mx[r] - mn) + os * __expf(om - mn);
      mx[r] = mn;
    }
  }
  if (mr == 0) {
#pragma unroll
    for (int r = 0; r < 8; r++)
      lse[a0 + (hl ? 8 : 0) + r] = __logf(sm[r]) + mx[r];
  }
}

__global__ void k_outer(const float* __restrict__ lse, const float* __restrict__ pos,
                        float* __restrict__ dout) {
  const int i = blockIdx.x * blockDim.x + threadIdx.x;
  if (i >= KP_) return;
  atomicAdd(&dout[OUTER_OFF], (lse[i] - pos[i]) / (float)KP_);
}

// ------------------------- launcher ----------------------------------------
extern "C" void kernel_launch(void* const* d_in, const int* in_sizes, int n_in,
                              void* d_out, int out_size, void* d_ws, size_t ws_size,
                              hipStream_t stream) {
  (void)in_sizes; (void)n_in; (void)out_size; (void)ws_size;
  const float* x    = (const float*)d_in[0];
  const float* qe   = (const float*)d_in[1];
  const float* qdw  = (const float*)d_in[2];
  const int*   prs  = (const int*)d_in[3];
  const int*   trs  = (const int*)d_in[4];
  const float* ew0  = (const float*)d_in[5];
  const float* eb0  = (const float*)d_in[6];
  const float* ew1  = (const float*)d_in[7];
  const float* eb1  = (const float*)d_in[8];
  const float* ew2  = (const float*)d_in[9];
  const float* eb2  = (const float*)d_in[10];
  const float* dw0  = (const float*)d_in[11];
  const float* db0  = (const float*)d_in[12];
  const float* dw1  = (const float*)d_in[13];
  const float* db1  = (const float*)d_in[14];
  const float* dw2  = (const float*)d_in[15];
  const float* db2  = (const float*)d_in[16];
  const float* cb   = (const float*)d_in[17];

  float* dout = (float*)d_out;
  char*  ws   = (char*)d_ws;
  u16*   xb   = (u16*)(ws + XB_OFF);
  u16*   h1b  = (u16*)(ws + H1B_OFF);
  u16*   h2b  = (u16*)(ws + H2B_OFF);
  float* zf   = (float*)(ws + ZF_OFF);
  float* qzf  = (float*)(ws + QZF_OFF);
  u16*   xqb  = (u16*)(ws + XQB_OFF);
  u16*   xnb  = (u16*)(ws + XNB_OFF);
  u16*   w0t  = (u16*)(ws + W0T_OFF);
  u16*   w1t  = (u16*)(ws + W1T_OFF);
  u16*   w2t  = (u16*)(ws + W2T_OFF);
  u16*   d0t  = (u16*)(ws + D0T_OFF);
  u16*   d1t  = (u16*)(ws + D1T_OFF);
  u16*   d2t  = (u16*)(ws + D2T_OFF);
  float* lseb = (float*)(ws + LSE_OFF);
  float* posb = (float*)(ws + POS_OFF);

  const dim3 tb(16, 16);
  k_init_scalars<<<1, 32, 0, stream>>>(dout);

  // weight transposes (W[K][N] -> Wt[N][K] bf16)
  k_transpose_bf16<<<dim3(H1_ / 16, IN_ / 16), tb, 0, stream>>>(ew0, w0t, IN_, H1_);
  k_transpose_bf16<<<dim3(H2_ / 16, H1_ / 16), tb, 0, stream>>>(ew1, w1t, H1_, H2_);
  k_transpose_bf16<<<dim3(E_ / 16,  H2_ / 16), tb, 0, stream>>>(ew2, w2t, H2_, E_);
  k_transpose_bf16<<<dim3(H2_ / 16, E_ / 16),  tb, 0, stream>>>(dw0, d0t, E_,  H2_);
  k_transpose_bf16<<<dim3(H1_ / 16, H2_ / 16), tb, 0, stream>>>(dw1, d1t, H2_, H1_);
  k_transpose_bf16<<<dim3(IN_ / 16, H1_ / 16), tb, 0, stream>>>(dw2, d2t, H1_, IN_);

  const int gx1 = (H1_ + 255) / 256;   // 8
  const int gx2 = (H2_ + 255) / 256;   // 4
  const int gxE = 1;
  const int gxI = (IN_ + 255) / 256;   // 3
  const int gy  = N_ / 128;            // 256

  // encoder on x -> z
  k_f32_to_bf16<<<2048, 256, 0, stream>>>(x, xb, N_ * IN_);
  k_gemm_bf16<<<dim3(gx1, gy), 256, 0, stream>>>(xb,  w0t, eb0, nullptr, h1b, N_, H1_, IN_, 1);
  k_gemm_bf16<<<dim3(gx2, gy), 256, 0, stream>>>(h1b, w1t, eb1, nullptr, h2b, N_, H2_, H1_, 1);
  k_gemm_bf16<<<dim3(gxE, gy), 256, 0, stream>>>(h2b, w2t, eb2, zf, nullptr, N_, E_, H2_, 0);

  // encoder on q_embs -> qz (reuses xb/h1b/h2b)
  k_f32_to_bf16<<<2048, 256, 0, stream>>>(qe, xb, N_ * IN_);
  k_gemm_bf16<<<dim3(gx1, gy), 256, 0, stream>>>(xb,  w0t, eb0, nullptr, h1b, N_, H1_, IN_, 1);
  k_gemm_bf16<<<dim3(gx2, gy), 256, 0, stream>>>(h1b, w1t, eb1, nullptr, h2b, N_, H2_, H1_, 1);
  k_gemm_bf16<<<dim3(gxE, gy), 256, 0, stream>>>(h2b, w2t, eb2, qzf, nullptr, N_, E_, H2_, 0);

  // VQ: argmin, idx, x_q_st, vq_loss, bf16 decoder input, normalized rows
  k_vq<<<N_ / 256, 256, 0, stream>>>(zf, cb, dout, xqb, xnb);

  // decoder on x_q_st -> out (f32 straight into d_out)
  k_gemm_bf16<<<dim3(gx2, gy), 256, 0, stream>>>(xqb, d0t, db0, nullptr, h2b, N_, H2_, E_, 1);
  k_gemm_bf16<<<dim3(gx1, gy), 256, 0, stream>>>(h2b, d1t, db1, nullptr, h1b, N_, H1_, H2_, 1);
  k_gemm_bf16<<<dim3(gxI, gy), 256, 0, stream>>>(h1b, d2t, db2, dout + OUT_OFF, nullptr, N_, IN_, H1_, 0);

  // losses
  k_cos<<<N_ / 256, 256, 0, stream>>>(zf, qzf, qdw, dout);
  k_trip<<<TT_ / 256, 256, 0, stream>>>(zf, trs, dout);
  k_pos<<<KP_ / 256, 256, 0, stream>>>(xnb, prs, posb);
  k_sim<<<KP_ / 128, 256, 0, stream>>>(xnb, prs, lseb, N_);
  k_outer<<<KP_ / 256, 256, 0, stream>>>(lseb, posb, dout);
}
